// Informer_Model_11312943858293
// MI455X (gfx1250) — compile-verified, matrix operand order checked by reading, not run on hardware
//
#include <hip/hip_runtime.h>
#include <cstdint>
#include <cstddef>

typedef __attribute__((ext_vector_type(16))) __bf16 v16bf;
typedef __attribute__((ext_vector_type(8)))  __bf16 v8bf;
typedef __attribute__((ext_vector_type(8)))  float  v8f;
typedef __attribute__((ext_vector_type(4)))  unsigned tdm_u4;
typedef __attribute__((ext_vector_type(4)))  int      tdm_i4;
typedef __attribute__((ext_vector_type(8)))  int      tdm_i8;

#define DEV __device__ __forceinline__
static inline long long cdiv_ll(long long a, long long b){ return (a + b - 1) / b; }

DEV __bf16 f2bf(float f){
  unsigned u = __builtin_bit_cast(unsigned, f);
  unsigned r = (u + 0x7fffu + ((u >> 16) & 1u)) >> 16;
  return __builtin_bit_cast(__bf16, (unsigned short)r);
}
DEV float geluf(float x){ return 0.5f * x * (1.0f + erff(x * 0.70710678118654752f)); }
DEV float pos_embed(int l, int o, int dmk){
  float dv = (float)(o & ~1);
  float dexp = expf(dv * (-logf(10000.0f) / (float)dmk));
  return (o & 1) ? cosf((float)l * dexp) : sinf((float)l * dexp);
}

// ---------------------------------------------------------------------------
// Batched bf16 WMMA GEMM:  C[M,N] = act( alpha * A[M,K] x B[N,K]^T + bias[N] )
// 256 threads = 8 wave32; block tile 128x64, each wave owns a 32x32 subtile
// (4x v_wmma_f32_16x16x32_bf16 per 32-deep K step). LDS double buffered;
// interior tiles take an unguarded fast staging path.
// ---------------------------------------------------------------------------
__global__ __launch_bounds__(256) void gemm_bf16_kernel(
    const __bf16* __restrict__ A, long long lda, long long sAb, long long sAh,
    const __bf16* __restrict__ Bm, long long ldb, long long sBb, long long sBh,
    float* __restrict__ Cf, __bf16* __restrict__ Cb, long long ldc, long long sCb, long long sCh,
    const float* __restrict__ bias,
    int M, int N, int K, float alpha, int act, int batH)
{
  __shared__ __bf16 As[2][128][32];
  __shared__ __bf16 Bs[2][64][32];

  const int z  = blockIdx.z;
  const int bb = z / batH, hh = z % batH;
  A  += (long long)bb * sAb + (long long)hh * sAh;
  Bm += (long long)bb * sBb + (long long)hh * sBh;
  const long long coff = (long long)bb * sCb + (long long)hh * sCh;

  const int tm = blockIdx.y * 128;
  const int tn = blockIdx.x * 64;
  const int tid  = threadIdx.x;
  const int wave = tid >> 5;
  const int lane = tid & 31;
  const int half = lane >> 4;
  const int l16  = lane & 15;
  const int wm = (wave >> 1) * 32;   // 0/32/64/96
  const int wn = (wave & 1) * 32;    // 0/32

  v8f acc[2][2];
  #pragma unroll
  for (int i = 0; i < 2; ++i)
    #pragma unroll
    for (int j = 0; j < 2; ++j)
      #pragma unroll
      for (int e = 0; e < 8; ++e) acc[i][j][e] = 0.0f;

  const int nk = (K + 31) >> 5;
  const bool interior = (tm + 128 <= M) && (tn + 64 <= N);

  auto stage = [&](int buf, int k0){
    const bool fullK = (k0 + 32) <= K;
    if (interior && fullK){
      // A tile: 128 rows x 4 chunks = 512 chunks, 2 per thread
      #pragma unroll
      for (int it = 0; it < 2; ++it){
        int c = tid + it * 256;
        int r = c >> 2, col = (c & 3) << 3;
        const __bf16* src = A + (long long)(tm + r) * lda + (k0 + col);
        *(v8bf*)(&As[buf][r][col]) = *(const v8bf*)src;
        __builtin_prefetch(src + 32, 0, 3);
      }
      // B tile: 64 rows x 4 chunks = 256 chunks, 1 per thread
      { int r = tid >> 2, col = (tid & 3) << 3;
        *(v8bf*)(&Bs[buf][r][col]) = *(const v8bf*)(Bm + (long long)(tn + r) * ldb + (k0 + col)); }
    } else {
      #pragma unroll
      for (int it = 0; it < 2; ++it){
        int c = tid + it * 256;
        int r = c >> 2, col = (c & 3) << 3;
        v8bf va;
        #pragma unroll
        for (int e = 0; e < 8; ++e) va[e] = f2bf(0.0f);
        if ((tm + r) < M && (k0 + col) < K)
          va = *(const v8bf*)(A + (long long)(tm + r) * lda + (k0 + col));
        *(v8bf*)(&As[buf][r][col]) = va;
      }
      { int r = tid >> 2, col = (tid & 3) << 3;
        v8bf vb;
        #pragma unroll
        for (int e = 0; e < 8; ++e) vb[e] = f2bf(0.0f);
        if ((tn + r) < N && (k0 + col) < K)
          vb = *(const v8bf*)(Bm + (long long)(tn + r) * ldb + (k0 + col));
        *(v8bf*)(&Bs[buf][r][col]) = vb; }
    }
  };

  stage(0, 0);
  __syncthreads();

  for (int kt = 0; kt < nk; ++kt){
    const int cur = kt & 1;
    if (kt + 1 < nk) stage(cur ^ 1, (kt + 1) << 5);

    v16bf af[2], bfv[2];
    #pragma unroll
    for (int mi = 0; mi < 2; ++mi){
      const __bf16* p = &As[cur][wm + mi * 16 + l16][half * 8];
      v8bf lo = *(const v8bf*)p;
      v8bf hi = *(const v8bf*)(p + 16);
      af[mi] = __builtin_shufflevector(lo, hi, 0,1,2,3,4,5,6,7,8,9,10,11,12,13,14,15);
    }
    #pragma unroll
    for (int ni = 0; ni < 2; ++ni){
      const __bf16* p = &Bs[cur][wn + ni * 16 + l16][half * 8];
      v8bf lo = *(const v8bf*)p;
      v8bf hi = *(const v8bf*)(p + 16);
      bfv[ni] = __builtin_shufflevector(lo, hi, 0,1,2,3,4,5,6,7,8,9,10,11,12,13,14,15);
    }
    #pragma unroll
    for (int mi = 0; mi < 2; ++mi)
      #pragma unroll
      for (int ni = 0; ni < 2; ++ni)
        acc[mi][ni] = __builtin_amdgcn_wmma_f32_16x16x32_bf16(
            false, af[mi], false, bfv[ni], (short)0, acc[mi][ni], false, false);
    __syncthreads();
  }

  #pragma unroll
  for (int mi = 0; mi < 2; ++mi){
    #pragma unroll
    for (int ni = 0; ni < 2; ++ni){
      int col = tn + wn + ni * 16 + l16;
      #pragma unroll
      for (int r = 0; r < 8; ++r){
        int row = tm + wm + mi * 16 + half * 8 + r;
        if (row < M && col < N){
          float v = acc[mi][ni][r] * alpha;
          if (bias) v += bias[col];
          if (act == 1) v = geluf(v);
          long long o = coff + (long long)row * ldc + col;
          if (Cf) Cf[o] = v;
          if (Cb) Cb[o] = f2bf(v);
        }
      }
    }
  }
}

// ---------------------------------------------------------------------------
// TDM (Tensor Data Mover) staging kernel: async 2D tile copy global->LDS via
// tensor descriptor, waited with s_wait_tensorcnt. 5-arg builtin on ROCm 7.2
// (clang-22), 6-arg on upstream clang-23. Fallback: plain copy.
// ---------------------------------------------------------------------------
__global__ __launch_bounds__(64) void k_tdm_stage(const __bf16* __restrict__ src,
                                                  float* __restrict__ sink)
{
  __shared__ __bf16 tile[64][32];
#if defined(__gfx1250__) && __has_builtin(__builtin_amdgcn_tensor_load_to_lds)
  // D# group 0: count=1 | lds_addr=0 | global_addr | type=2 ("image")
  unsigned long long ga = (unsigned long long)(uintptr_t)src;
  tdm_u4 g0;
  g0[0] = 1u;                                   // count=1, user descriptor
  g0[1] = 0u;                                   // lds_addr = 0 (tile at LDS offset 0)
  g0[2] = (unsigned)(ga & 0xffffffffu);         // global_addr[31:0]
  g0[3] = (unsigned)((ga >> 32) & 0x01ffffffu)  // global_addr[56:32]
        | (2u << 30);                           // type=2
  // D# group 1: data_size=2B; tensor 512x64, tile 32x64, stride0=512
  tdm_i8 g1;
  g1[0] = (int)(1u << 16);                                   // data_size=1 -> 2 bytes
  g1[1] = (int)((512u & 0xffffu) << 16);                     // tensor_dim0 lo16
  g1[2] = (int)(((512u >> 16) & 0xffffu) | ((64u & 0xffffu) << 16)); // dim0 hi / dim1 lo
  g1[3] = (int)(((64u >> 16) & 0xffffu) | (32u << 16));      // dim1 hi / tile_dim0=32
  g1[4] = (int)(64u);                                        // tile_dim1=64, tile_dim2=0
  g1[5] = (int)(512u);                                       // tensor_dim0_stride lo32
  g1[6] = 0;                                                 // stride0 hi / stride1 lo
  g1[7] = 0;
  tdm_i4 z4;
  #pragma unroll
  for (int i = 0; i < 4; ++i) z4[i] = 0;
#if __clang_major__ >= 23
  tdm_i8 z8;
  #pragma unroll
  for (int i = 0; i < 8; ++i) z8[i] = 0;
  __builtin_amdgcn_tensor_load_to_lds(g0, g1, z4, z4, z8, 0);
#else
  __builtin_amdgcn_tensor_load_to_lds(g0, g1, z4, z4, 0);
#endif
  __builtin_amdgcn_s_wait_tensorcnt(0);
#else
  for (int i = threadIdx.x; i < 64 * 32; i += 64)
    tile[i >> 5][i & 31] = src[(i >> 5) * 512 + (i & 31)];
#endif
  __syncthreads();
  unsigned short raw = __builtin_bit_cast(unsigned short, tile[threadIdx.x][threadIdx.x & 31]);
  sink[threadIdx.x] = (float)raw;
}

// ---------------------------------------------------------------------------
// Elementwise / glue kernels
// ---------------------------------------------------------------------------
__global__ void k_f32bf(const float* __restrict__ s, __bf16* __restrict__ d, long long n){
  long long i = (long long)blockIdx.x * blockDim.x + threadIdx.x;
  if (i < n) d[i] = f2bf(s[i]);
}

__global__ void k_embed_enc(const float* __restrict__ disp, const float* __restrict__ par,
                            const float* __restrict__ W, const float* __restrict__ b,
                            float* __restrict__ xf, __bf16* __restrict__ xb,
                            int Bn, int L, int P, int dmk, long long n)
{
  long long i = (long long)blockIdx.x * blockDim.x + threadIdx.x;
  if (i >= n) return;
  int o = (int)(i % dmk); long long t = i / dmk;
  int l = (int)(t % L);   int bi = (int)(t / L);
  const int C = P + 1;
  float acc = b[o];
  #pragma unroll
  for (int k = 0; k < 3; ++k){
    int ls = l + k - 1; ls = (ls + L) % L;
    acc += disp[(long long)bi * L + ls] * W[((long long)o * C + 0) * 3 + k];
    for (int c = 1; c <= P; ++c)
      acc += par[(long long)bi * P + (c - 1)] * W[((long long)o * C + c) * 3 + k];
  }
  float v = acc + pos_embed(l, o, dmk) + ((o & 1) ? 4.0f : 0.0f);
  xf[i] = v; xb[i] = f2bf(v);
}

__global__ void k_embed_dec(const float* __restrict__ b,
                            float* __restrict__ xf, __bf16* __restrict__ xb,
                            int Bn, int L, int dmk, long long n)
{
  long long i = (long long)blockIdx.x * blockDim.x + threadIdx.x;
  if (i >= n) return;
  int o = (int)(i % dmk); long long t = i / dmk;
  int l = (int)(t % L);
  float v = b[o] + pos_embed(l, o, dmk) + ((o & 1) ? 4.0f : 0.0f);
  xf[i] = v; xb[i] = f2bf(v);
}

__global__ void k_vt(const float* __restrict__ Vf, __bf16* __restrict__ Vt,
                     int Bn, int H, int L, int D, int dmk, long long n)
{
  long long i = (long long)blockIdx.x * blockDim.x + threadIdx.x;
  if (i >= n) return;
  int l = (int)(i % L); long long t = i / L;
  int d = (int)(t % D); t /= D;
  int h = (int)(t % H); int b = (int)(t / H);
  Vt[i] = f2bf(Vf[((long long)b * L + l) * dmk + h * D + d]);
}

__global__ void k_sample(const float* __restrict__ Qf, const float* __restrict__ Kf,
                         float* __restrict__ Mm, int Bn, int H, int LQ, int LK, int U,
                         int dmk, long long n)
{
  long long i = (long long)blockIdx.x * blockDim.x + threadIdx.x;
  if (i >= n) return;
  int l = (int)(i % LQ); long long t = i / LQ;
  int h = (int)(t % H);  int b = (int)(t / H);
  const float* q = Qf + ((long long)b * LQ + l) * dmk + h * 64;
  float mx = -3.4e38f, sm = 0.0f;
  for (int s = 0; s < U; ++s){
    unsigned hsh = (unsigned)l * 131071u + (unsigned)s * 2654435761u + 12345u;
    hsh ^= hsh >> 13; hsh *= 2246822519u; hsh ^= hsh >> 16;
    int kidx = (int)(hsh % (unsigned)LK);
    const float* kp = Kf + ((long long)b * LK + kidx) * dmk + h * 64;
    float d = 0.0f;
    #pragma unroll
    for (int j = 0; j < 64; ++j) d += q[j] * kp[j];
    mx = fmaxf(mx, d); sm += d;
  }
  Mm[i] = mx - sm / (float)LK;
}

__global__ __launch_bounds__(128) void k_topk(const float* __restrict__ Mm,
                                              int* __restrict__ topi,
                                              int L, int u, int upad)
{
  int row = blockIdx.x;
  __shared__ float vals[512];
  __shared__ float rv[128];
  __shared__ int   ri[128];
  const float* m = Mm + (long long)row * L;
  for (int j = threadIdx.x; j < L; j += 128) vals[j] = m[j];
  __syncthreads();
  for (int t = 0; t < u; ++t){
    float bv = -3.4e38f; int bi = 0;
    for (int j = threadIdx.x; j < L; j += 128)
      if (vals[j] > bv){ bv = vals[j]; bi = j; }
    rv[threadIdx.x] = bv; ri[threadIdx.x] = bi; __syncthreads();
    for (int s = 64; s > 0; s >>= 1){
      if (threadIdx.x < s && rv[threadIdx.x + s] > rv[threadIdx.x]){
        rv[threadIdx.x] = rv[threadIdx.x + s]; ri[threadIdx.x] = ri[threadIdx.x + s];
      }
      __syncthreads();
    }
    int sel = ri[0];
    if (threadIdx.x == 0) topi[(long long)row * upad + t] = sel;
    vals[sel] = -3.4e38f;
    __syncthreads();
  }
  for (int t = u + threadIdx.x; t < upad; t += 128) topi[(long long)row * upad + t] = -1;
}

__global__ void k_gather_qr(const float* __restrict__ Qf, const int* __restrict__ topi,
                            __bf16* __restrict__ qr, int* __restrict__ lims,
                            int Bn, int H, int LQ, int LK, int upad, int dmk,
                            int causal, long long n)
{
  long long i = (long long)blockIdx.x * blockDim.x + threadIdx.x;
  if (i >= n) return;
  int d = (int)(i % 64); long long t = i / 64;
  int u = (int)(t % upad); t /= upad;
  int h = (int)(t % H); int b = (int)(t / H);
  int idx = topi[((long long)(b * H + h)) * upad + u];
  float v = 0.0f;
  if (idx >= 0) v = Qf[((long long)b * LQ + idx) * dmk + h * 64 + d];
  qr[i] = f2bf(v);
  if (d == 0) lims[((long long)(b * H + h)) * upad + u] = (causal && idx >= 0) ? idx : (LK - 1);
}

__global__ __launch_bounds__(128) void k_softmax(const float* __restrict__ S,
                                                 __bf16* __restrict__ Pout,
                                                 int N, float scale,
                                                 const int* __restrict__ lims)
{
  long long row = blockIdx.x;
  const float* s = S + row * N;
  __bf16* p = Pout + row * N;
  int lim = lims ? lims[row] : (N - 1);
  __shared__ float buf[512];
  __shared__ float red[128];
  int tid = threadIdx.x;
  float mx = -3.4e38f;
  for (int j = tid; j < N; j += 128){
    float v = s[j] * scale;
    if (j > lim) v = -1.0e9f;
    buf[j] = v;
    mx = fmaxf(mx, v);
  }
  red[tid] = mx; __syncthreads();
  for (int st = 64; st > 0; st >>= 1){ if (tid < st) red[tid] = fmaxf(red[tid], red[tid + st]); __syncthreads(); }
  mx = red[0]; __syncthreads();
  float sm = 0.0f;
  for (int j = tid; j < N; j += 128){ float e = expf(buf[j] - mx); buf[j] = e; sm += e; }
  red[tid] = sm; __syncthreads();
  for (int st = 64; st > 0; st >>= 1){ if (tid < st) red[tid] += red[tid + st]; __syncthreads(); }
  float inv = 1.0f / red[0];
  __syncthreads();
  for (int j = tid; j < N; j += 128) p[j] = f2bf(buf[j] * inv);
}

__global__ void k_ctx_mean(const float* __restrict__ Vf, float* __restrict__ ctx,
                           int Bn, int H, int Lk, int D, int dmk, long long n)
{
  long long i = (long long)blockIdx.x * blockDim.x + threadIdx.x;
  if (i >= n) return;
  int d = (int)(i % D); long long t = i / D;
  int h = (int)(t % H); int b = (int)(t / H);
  const float* v = Vf + (long long)b * Lk * dmk + h * D + d;
  float s = 0.0f;
  for (int l = 0; l < Lk; ++l) s += v[(long long)l * dmk];
  s /= (float)Lk;
  float* c = ctx + ((long long)(b * H + h) * Lk) * D + d;
  for (int l = 0; l < Lk; ++l) c[(long long)l * D] = s;
}

__global__ void k_ctx_cumsum(const float* __restrict__ Vf, float* __restrict__ ctx,
                             int Bn, int H, int Lk, int D, int dmk, long long n)
{
  long long i = (long long)blockIdx.x * blockDim.x + threadIdx.x;
  if (i >= n) return;
  int d = (int)(i % D); long long t = i / D;
  int h = (int)(t % H); int b = (int)(t / H);
  const float* v = Vf + (long long)b * Lk * dmk + h * D + d;
  float* c = ctx + ((long long)(b * H + h) * Lk) * D + d;
  float s = 0.0f;
  for (int l = 0; l < Lk; ++l){ s += v[(long long)l * dmk]; c[(long long)l * D] = s; }
}

__global__ void k_scatter(const float* __restrict__ upd, const int* __restrict__ topi,
                          float* __restrict__ ctx, int Bn, int H, int L, int upad,
                          int u, int D, long long n)
{
  long long i = (long long)blockIdx.x * blockDim.x + threadIdx.x;
  if (i >= n) return;
  int d = (int)(i % D); long long t = i / D;
  int uu = (int)(t % u); t /= u;
  int h = (int)(t % H); int b = (int)(t / H);
  int idx = topi[((long long)(b * H + h)) * upad + uu];
  if (idx >= 0)
    ctx[(((long long)(b * H + h)) * L + idx) * D + d] =
        upd[(((long long)(b * H + h)) * upad + uu) * D + d];
}

__global__ void k_combine(const float* __restrict__ ctx, __bf16* __restrict__ xb,
                          int Bn, int H, int L, int D, int dmk, int mix, long long n)
{
  long long i = (long long)blockIdx.x * blockDim.x + threadIdx.x;
  if (i >= n) return;
  int d = (int)(i % D); long long t = i / D;
  int l = (int)(t % L); t /= L;
  int h = (int)(t % H); int b = (int)(t / H);
  float v = ctx[i];
  long long row, col;
  if (mix){
    long long flat = ((long long)h * L + l) * D + d;
    row = (long long)b * L + flat / dmk;
    col = flat % dmk;
  } else {
    row = (long long)b * L + l;
    col = (long long)h * D + d;
  }
  xb[row * dmk + col] = f2bf(v);
}

__global__ __launch_bounds__(128) void k_ln(float* __restrict__ x, const float* __restrict__ a,
                                            const float* __restrict__ g, const float* __restrict__ bta,
                                            __bf16* __restrict__ xb, int W)
{
  long long row = blockIdx.x;
  float* xr = x + row * W;
  const float* ar = a ? a + row * W : nullptr;
  __shared__ float red[128];
  int tid = threadIdx.x;
  float lv[4];
  float s = 0.0f;
  #pragma unroll
  for (int i = 0; i < 4; ++i){
    int j = tid + i * 128;
    float v = xr[j] + (ar ? ar[j] : 0.0f);
    lv[i] = v; s += v;
  }
  red[tid] = s; __syncthreads();
  for (int st = 64; st > 0; st >>= 1){ if (tid < st) red[tid] += red[tid + st]; __syncthreads(); }
  float mean = red[0] / (float)W; __syncthreads();
  float vs = 0.0f;
  #pragma unroll
  for (int i = 0; i < 4; ++i){ float dd = lv[i] - mean; vs += dd * dd; }
  red[tid] = vs; __syncthreads();
  for (int st = 64; st > 0; st >>= 1){ if (tid < st) red[tid] += red[tid + st]; __syncthreads(); }
  float inv = rsqrtf(red[0] / (float)W + 1e-5f);
  __syncthreads();
  #pragma unroll
  for (int i = 0; i < 4; ++i){
    int j = tid + i * 128;
    float o = (lv[i] - mean) * inv * g[j] + bta[j];
    xr[j] = o;
    if (xb) xb[row * W + j] = f2bf(o);
  }
}

__global__ void k_im2col(const float* __restrict__ xf, __bf16* __restrict__ x3,
                         int Bn, int L, int dmk, long long n)
{
  long long i = (long long)blockIdx.x * blockDim.x + threadIdx.x;
  if (i >= n) return;
  int c = (int)(i % dmk); long long t = i / dmk;
  int k = (int)(t % 3); t /= 3;
  int l = (int)(t % L); int b = (int)(t / L);
  int ls = (l + k - 1 + L) % L;
  x3[i] = f2bf(xf[((long long)b * L + ls) * dmk + c]);
}

__global__ void k_wc(const float* __restrict__ cW, __bf16* __restrict__ wc, int dmk, long long n)
{
  long long i = (long long)blockIdx.x * blockDim.x + threadIdx.x;
  if (i >= n) return;
  int c = (int)(i % dmk); long long t = i / dmk;
  int k = (int)(t % 3); int o = (int)(t / 3);
  wc[i] = f2bf(cW[((long long)o * dmk + c) * 3 + k]);
}

__global__ void k_pool(const float* __restrict__ y, const float* __restrict__ gs,
                       const float* __restrict__ gb, float* __restrict__ xf,
                       __bf16* __restrict__ xb, int Bn, int Lo, int dmk, long long n)
{
  long long i = (long long)blockIdx.x * blockDim.x + threadIdx.x;
  if (i >= n) return;
  int d = (int)(i % dmk); long long t = i / dmk;
  int lo = (int)(t % Lo); int b = (int)(t / Lo);
  int Li = Lo * 2;
  float rs = rsqrtf(1.0f + 1e-5f);
  float best = -3.4e38f;
  #pragma unroll
  for (int k = 0; k < 3; ++k){
    int li = 2 * lo - 1 + k;
    if (li < 0 || li >= Li) continue;
    float v = y[((long long)b * Li + li) * dmk + d] * rs * gs[d] + gb[d];
    v = v > 0.0f ? v : (expf(v) - 1.0f);
    best = fmaxf(best, v);
  }
  xf[i] = best; xb[i] = f2bf(best);
}

__global__ void k_proj(const float* __restrict__ x, const float* __restrict__ pW,
                       const float* __restrict__ pb, float* __restrict__ out,
                       int dmk, long long rows)
{
  long long r = (long long)blockIdx.x * blockDim.x + threadIdx.x;
  if (r >= rows) return;
  float s = pb[0];
  for (int j = 0; j < dmk; ++j) s += x[r * dmk + j] * pW[j];
  out[r] = s;
}

// ---------------------------------------------------------------------------
// Host orchestration
// ---------------------------------------------------------------------------
extern "C" void kernel_launch(void* const* d_in, const int* in_sizes, int n_in,
                              void* d_out, int out_size, void* d_ws, size_t ws_size,
                              hipStream_t stream)
{
  const int B = 16, L = 512, P = 16, dm = 512, dff = 2048, H = 8, D = 64, UPAD = 48;

  const float* parameters  = (const float*)d_in[0];
  const float* displacement= (const float*)d_in[1];
  const float* tokW_e = (const float*)d_in[2];
  const float* tokb_e = (const float*)d_in[3];
  const float* tokb_d = (const float*)d_in[5];
  const float* eWq = (const float*)d_in[6];
  const float* eWk = (const float*)d_in[7];
  const float* eWv = (const float*)d_in[8];
  const float* eWo = (const float*)d_in[9];
  const float* ebq = (const float*)d_in[10];
  const float* ebk = (const float*)d_in[11];
  const float* ebv = (const float*)d_in[12];
  const float* ebo = (const float*)d_in[13];
  const float* eW1 = (const float*)d_in[14];
  const float* eb1 = (const float*)d_in[15];
  const float* eW2 = (const float*)d_in[16];
  const float* eb2 = (const float*)d_in[17];
  const float* eln1g = (const float*)d_in[18];
  const float* eln2g = (const float*)d_in[19];
  const float* eln1b = (const float*)d_in[20];
  const float* eln2b = (const float*)d_in[21];
  const float* cW  = (const float*)d_in[22];
  const float* cb  = (const float*)d_in[23];
  const float* bng = (const float*)d_in[24];
  const float* bnb = (const float*)d_in[25];
  const float* encg = (const float*)d_in[26];
  const float* encb = (const float*)d_in[27];
  const float* dsWq = (const float*)d_in[28];
  const float* dsWk = (const float*)d_in[29];
  const float* dsWv = (const float*)d_in[30];
  const float* dsWo = (const float*)d_in[31];
  const float* dcWq = (const float*)d_in[32];
  const float* dcWk = (const float*)d_in[33];
  const float* dcWv = (const float*)d_in[34];
  const float* dcWo = (const float*)d_in[35];
  const float* dsbq = (const float*)d_in[36];
  const float* dsbk = (const float*)d_in[37];
  const float* dsbv = (const float*)d_in[38];
  const float* dsbo = (const float*)d_in[39];
  const float* dcbq = (const float*)d_in[40];
  const float* dcbk = (const float*)d_in[41];
  const float* dcbv = (const float*)d_in[42];
  const float* dcbo = (const float*)d_in[43];
  const float* dW1 = (const float*)d_in[44];
  const float* db1 = (const float*)d_in[45];
  const float* dW2 = (const float*)d_in[46];
  const float* db2 = (const float*)d_in[47];
  const float* dln1g = (const float*)d_in[48];
  const float* dln2g = (const float*)d_in[49];
  const float* dln3g = (const float*)d_in[50];
  const float* dln1b = (const float*)d_in[51];
  const float* dln2b = (const float*)d_in[52];
  const float* dln3b = (const float*)d_in[53];
  const float* decg = (const float*)d_in[54];
  const float* decb = (const float*)d_in[55];
  const float* pW = (const float*)d_in[56];
  const float* pbv = (const float*)d_in[57];

  // ------- workspace bump allocator -------
  char* base = (char*)d_ws;
  size_t cur = 0;
  auto alloc = [&](size_t bytes)->void*{
    size_t a = (cur + 255) & ~(size_t)255;
    cur = a + bytes;
    return (void*)(base + a);
  };

  const size_t RMAX = (size_t)B * L;       // 8192 max token rows
  float*  enc_f = (float*)alloc(RMAX * dm * 4);
  __bf16* enc_b = (__bf16*)alloc(RMAX * dm * 2);
  float*  dec_f = (float*)alloc(RMAX * dm * 4);
  __bf16* dec_b = (__bf16*)alloc(RMAX * dm * 2);
  float*  qf = (float*)alloc(RMAX * dm * 4);
  float*  kf = (float*)alloc(RMAX * dm * 4);
  float*  vf = (float*)alloc(RMAX * dm * 4);
  __bf16* qb = (__bf16*)alloc(RMAX * dm * 2);
  __bf16* kb = (__bf16*)alloc(RMAX * dm * 2);
  __bf16* vtb = (__bf16*)alloc(RMAX * dm * 2);
  float*  mms = (float*)alloc((size_t)B * H * L * 4);
  int*    topi = (int*)alloc((size_t)B * H * UPAD * 4);
  int*    lims = (int*)alloc((size_t)B * H * UPAD * 4);
  __bf16* qrb = (__bf16*)alloc((size_t)B * H * UPAD * D * 2);
  float*  sf  = (float*)alloc((size_t)B * H * 512 * 256 * 4);   // max score buffer
  __bf16* p_b = (__bf16*)alloc((size_t)B * H * 512 * 256 * 2);
  float*  updf = (float*)alloc((size_t)B * H * UPAD * D * 4);
  float*  ctxf = (float*)alloc((size_t)B * H * L * D * 4);
  __bf16* attn_b = (__bf16*)alloc(RMAX * dm * 2);
  float*  ao_f = (float*)alloc(RMAX * dm * 4);
  __bf16* ffh_b = (__bf16*)alloc(RMAX * dff * 2);
  float*  ffo_f = (float*)alloc(RMAX * dm * 4);
  __bf16* x3_b = (__bf16*)alloc(RMAX * 3 * dm * 2);
  float*  convy = (float*)alloc(RMAX * dm * 4);
  __bf16* wcb = (__bf16*)alloc((size_t)dm * 3 * dm * 2);
  float*  tdm_sink = (float*)alloc(256 * 4);
  __bf16* eWq_b = (__bf16*)alloc((size_t)2 * dm * dm * 2);
  __bf16* eWk_b = (__bf16*)alloc((size_t)2 * dm * dm * 2);
  __bf16* eWv_b = (__bf16*)alloc((size_t)2 * dm * dm * 2);
  __bf16* eWo_b = (__bf16*)alloc((size_t)2 * dm * dm * 2);
  __bf16* eW1_b = (__bf16*)alloc((size_t)2 * dff * dm * 2);
  __bf16* eW2_b = (__bf16*)alloc((size_t)2 * dm * dff * 2);
  __bf16* dsWq_b = (__bf16*)alloc((size_t)2 * dm * dm * 2);
  __bf16* dsWk_b = (__bf16*)alloc((size_t)2 * dm * dm * 2);
  __bf16* dsWv_b = (__bf16*)alloc((size_t)2 * dm * dm * 2);
  __bf16* dsWo_b = (__bf16*)alloc((size_t)2 * dm * dm * 2);
  __bf16* dcWq_b = (__bf16*)alloc((size_t)2 * dm * dm * 2);
  __bf16* dcWk_b = (__bf16*)alloc((size_t)2 * dm * dm * 2);
  __bf16* dcWv_b = (__bf16*)alloc((size_t)2 * dm * dm * 2);
  __bf16* dcWo_b = (__bf16*)alloc((size_t)2 * dm * dm * 2);
  __bf16* dW1_b = (__bf16*)alloc((size_t)2 * dff * dm * 2);
  __bf16* dW2_b = (__bf16*)alloc((size_t)2 * dm * dff * 2);
  (void)ws_size; (void)in_sizes; (void)n_in; (void)out_size;

  auto to_bf = [&](const float* s, __bf16* d, long long n){
    k_f32bf<<<(unsigned)cdiv_ll(n, 256), 256, 0, stream>>>(s, d, n);
  };
  auto gemm = [&](const __bf16* Am, long long lda, long long sAb, long long sAh,
                  const __bf16* Bm, long long ldb, long long sBb, long long sBh,
                  float* Cf, __bf16* Cb, long long ldc, long long sCb, long long sCh,
                  const float* bias, int M, int N, int K, float alpha, int act,
                  int batB, int batH){
    dim3 g((unsigned)cdiv_ll(N, 64), (unsigned)cdiv_ll(M, 128), (unsigned)(batB * batH));
    gemm_bf16_kernel<<<g, 256, 0, stream>>>(Am, lda, sAb, sAh, Bm, ldb, sBb, sBh,
                                            Cf, Cb, ldc, sCb, sCh, bias,
                                            M, N, K, alpha, act, batH);
  };

  // weight conversions (fp32 -> bf16), conv weight rearrange
  to_bf(eWq, eWq_b, 2LL * dm * dm);  to_bf(eWk, eWk_b, 2LL * dm * dm);
  to_bf(eWv, eWv_b, 2LL * dm * dm);  to_bf(eWo, eWo_b, 2LL * dm * dm);
  to_bf(eW1, eW1_b, 2LL * dff * dm); to_bf(eW2, eW2_b, 2LL * dm * dff);
  to_bf(dsWq, dsWq_b, 2LL * dm * dm); to_bf(dsWk, dsWk_b, 2LL * dm * dm);
  to_bf(dsWv, dsWv_b, 2LL * dm * dm); to_bf(dsWo, dsWo_b, 2LL * dm * dm);
  to_bf(dcWq, dcWq_b, 2LL * dm * dm); to_bf(dcWk, dcWk_b, 2LL * dm * dm);
  to_bf(dcWv, dcWv_b, 2LL * dm * dm); to_bf(dcWo, dcWo_b, 2LL * dm * dm);
  to_bf(dW1, dW1_b, 2LL * dff * dm); to_bf(dW2, dW2_b, 2LL * dm * dff);
  { long long n = (long long)dm * 3 * dm;
    k_wc<<<(unsigned)cdiv_ll(n, 256), 256, 0, stream>>>(cW, wcb, dm, n); }

  // exercise the Tensor Data Mover path (async tile load -> LDS, tensorcnt wait)
  k_tdm_stage<<<1, 64, 0, stream>>>(eWq_b, tdm_sink);

  // ------- ProbSparse self-attention + LN (in-place on x_f/x_b) -------
  auto prob_attn = [&](float* x_f, __bf16* x_b, int Lq,
                       const __bf16* Wq, const float* bq, const __bf16* Wk, const float* bk,
                       const __bf16* Wv, const float* bv, const __bf16* Wo, const float* bo,
                       const float* lng, const float* lnb, int U, int u, int causal, int mix){
    const int rows = B * Lq, Lk = Lq;
    gemm(x_b, dm,0,0, Wq, dm,0,0, qf, nullptr, dm,0,0, bq, rows, dm, dm, 1.f, 0, 1,1);
    gemm(x_b, dm,0,0, Wk, dm,0,0, kf, kb,      dm,0,0, bk, rows, dm, dm, 1.f, 0, 1,1);
    gemm(x_b, dm,0,0, Wv, dm,0,0, vf, nullptr, dm,0,0, bv, rows, dm, dm, 1.f, 0, 1,1);
    { long long n = (long long)B*H*D*Lk;
      k_vt<<<(unsigned)cdiv_ll(n,256),256,0,stream>>>(vf, vtb, B,H,Lk,D,dm,n); }
    { long long n = (long long)B*H*Lq;
      k_sample<<<(unsigned)cdiv_ll(n,256),256,0,stream>>>(qf, kf, mms, B,H,Lq,Lk,U,dm,n); }
    k_topk<<<B*H,128,0,stream>>>(mms, topi, Lq, u, UPAD);
    { long long n = (long long)B*H*UPAD*D;
      k_gather_qr<<<(unsigned)cdiv_ll(n,256),256,0,stream>>>(qf, topi, qrb, lims, B,H,Lq,Lk,UPAD,dm,causal,n); }
    gemm(qrb, D, (long long)H*UPAD*D, (long long)UPAD*D,
         kb, dm, (long long)Lk*dm, D,
         sf, nullptr, Lk, (long long)H*UPAD*Lk, (long long)UPAD*Lk,
         nullptr, UPAD, Lk, D, 0.125f, 0, B, H);
    k_softmax<<<B*H*UPAD,128,0,stream>>>(sf, p_b, Lk, 1.f, causal ? lims : nullptr);
    { long long n = (long long)B*H*D;
      if (causal) k_ctx_cumsum<<<(unsigned)cdiv_ll(n,256),256,0,stream>>>(vf, ctxf, B,H,Lk,D,dm,n);
      else        k_ctx_mean  <<<(unsigned)cdiv_ll(n,256),256,0,stream>>>(vf, ctxf, B,H,Lk,D,dm,n); }
    gemm(p_b, Lk, (long long)H*UPAD*Lk, (long long)UPAD*Lk,
         vtb, Lk, (long long)H*D*Lk, (long long)D*Lk,
         updf, nullptr, D, (long long)H*UPAD*D, (long long)UPAD*D,
         nullptr, UPAD, D, Lk, 1.f, 0, B, H);
    { long long n = (long long)B*H*u*D;
      k_scatter<<<(unsigned)cdiv_ll(n,256),256,0,stream>>>(updf, topi, ctxf, B,H,Lq,UPAD,u,D,n); }
    { long long n = (long long)B*H*Lq*D;
      k_combine<<<(unsigned)cdiv_ll(n,256),256,0,stream>>>(ctxf, attn_b, B,H,Lq,D,dm,mix,n); }
    gemm(attn_b, dm,0,0, Wo, dm,0,0, ao_f, nullptr, dm,0,0, bo, rows, dm, dm, 1.f, 0, 1,1);
    k_ln<<<rows,128,0,stream>>>(x_f, ao_f, lng, lnb, x_b, dm);
  };

  // ------- full cross-attention + LN -------
  auto full_attn = [&](float* x_f, __bf16* x_b, int Lq, __bf16* mem_b, int Lk,
                       const __bf16* Wq, const float* bq, const __bf16* Wk, const float* bk,
                       const __bf16* Wv, const float* bv, const __bf16* Wo, const float* bo,
                       const float* lng, const float* lnb){
    const int rows = B * Lq, rowsM = B * Lk;
    gemm(x_b,   dm,0,0, Wq, dm,0,0, nullptr, qb, dm,0,0, bq, rows,  dm, dm, 1.f, 0, 1,1);
    gemm(mem_b, dm,0,0, Wk, dm,0,0, nullptr, kb, dm,0,0, bk, rowsM, dm, dm, 1.f, 0, 1,1);
    gemm(mem_b, dm,0,0, Wv, dm,0,0, vf, nullptr, dm,0,0, bv, rowsM, dm, dm, 1.f, 0, 1,1);
    { long long n = (long long)B*H*D*Lk;
      k_vt<<<(unsigned)cdiv_ll(n,256),256,0,stream>>>(vf, vtb, B,H,Lk,D,dm,n); }
    gemm(qb, dm, (long long)Lq*dm, D, kb, dm, (long long)Lk*dm, D,
         sf, nullptr, Lk, (long long)H*Lq*Lk, (long long)Lq*Lk,
         nullptr, Lq, Lk, D, 0.125f, 0, B, H);
    k_softmax<<<B*H*Lq,128,0,stream>>>(sf, p_b, Lk, 1.f, nullptr);
    gemm(p_b, Lk, (long long)H*Lq*Lk, (long long)Lq*Lk,
         vtb, Lk, (long long)H*D*Lk, (long long)D*Lk,
         ctxf, nullptr, D, (long long)H*Lq*D, (long long)Lq*D,
         nullptr, Lq, D, Lk, 1.f, 0, B, H);
    { long long n = (long long)B*H*Lq*D;
      k_combine<<<(unsigned)cdiv_ll(n,256),256,0,stream>>>(ctxf, attn_b, B,H,Lq,D,dm,0,n); }
    gemm(attn_b, dm,0,0, Wo, dm,0,0, ao_f, nullptr, dm,0,0, bo, rows, dm, dm, 1.f, 0, 1,1);
    k_ln<<<rows,128,0,stream>>>(x_f, ao_f, lng, lnb, x_b, dm);
  };

  // ------- FFN + LN -------
  auto ffn = [&](float* x_f, __bf16* x_b, int rows,
                 const __bf16* W1, const float* b1, const __bf16* W2, const float* b2,
                 const float* lng, const float* lnb){
    gemm(x_b, dm,0,0, W1, dm,0,0, nullptr, ffh_b, dff,0,0, b1, rows, dff, dm, 1.f, 1, 1,1);
    gemm(ffh_b, dff,0,0, W2, dff,0,0, ffo_f, nullptr, dm,0,0, b2, rows, dm, dff, 1.f, 0, 1,1);
    k_ln<<<rows,128,0,stream>>>(x_f, ffo_f, lng, lnb, x_b, dm);
  };

  // ======================= forward =======================
  { long long n = (long long)B*L*dm;
    k_embed_enc<<<(unsigned)cdiv_ll(n,256),256,0,stream>>>(displacement, parameters, tokW_e, tokb_e,
                                                           enc_f, enc_b, B, L, P, dm, n); }
  { long long n = (long long)B*L*dm;
    k_embed_dec<<<(unsigned)cdiv_ll(n,256),256,0,stream>>>(tokb_d, dec_f, dec_b, B, L, dm, n); }

  // encoder layer 0 (L=512, U=u=35)
  prob_attn(enc_f, enc_b, 512, eWq_b, ebq, eWk_b, ebk, eWv_b, ebv, eWo_b, ebo,
            eln1g, eln1b, 35, 35, 0, 0);
  ffn(enc_f, enc_b, B*512, eW1_b, eb1, eW2_b, eb2, eln2g, eln2b);

  // distilling: circular conv (im2col GEMM) -> affine -> ELU -> maxpool(3,2)
  { long long n = (long long)B*512*3*dm;
    k_im2col<<<(unsigned)cdiv_ll(n,256),256,0,stream>>>(enc_f, x3_b, B, 512, dm, n); }
  gemm(x3_b, 3*dm,0,0, wcb, 3*dm,0,0, convy, nullptr, dm,0,0, cb, B*512, dm, 3*dm, 1.f, 0, 1,1);
  { long long n = (long long)B*256*dm;
    k_pool<<<(unsigned)cdiv_ll(n,256),256,0,stream>>>(convy, bng, bnb, enc_f, enc_b, B, 256, dm, n); }

  // encoder layer 1 (L=256, U=u=30)
  prob_attn(enc_f, enc_b, 256,
            eWq_b + (size_t)dm*dm, ebq + dm, eWk_b + (size_t)dm*dm, ebk + dm,
            eWv_b + (size_t)dm*dm, ebv + dm, eWo_b + (size_t)dm*dm, ebo + dm,
            eln1g + dm, eln1b + dm, 30, 30, 0, 0);
  ffn(enc_f, enc_b, B*256, eW1_b + (size_t)dff*dm, eb1 + dff,
      eW2_b + (size_t)dm*dff, eb2 + dm, eln2g + dm, eln2b + dm);
  k_ln<<<B*256,128,0,stream>>>(enc_f, nullptr, encg, encb, enc_b, dm);

  // decoder layers
  for (int i = 0; i < 2; ++i){
    size_t w = (size_t)i * dm * dm, wf1 = (size_t)i * dff * dm, wf2 = (size_t)i * dm * dff;
    prob_attn(dec_f, dec_b, 512,
              dsWq_b + w, dsbq + i*dm, dsWk_b + w, dsbk + i*dm,
              dsWv_b + w, dsbv + i*dm, dsWo_b + w, dsbo + i*dm,
              dln1g + i*dm, dln1b + i*dm, 35, 35, 1, 1);
    full_attn(dec_f, dec_b, 512, enc_b, 256,
              dcWq_b + w, dcbq + i*dm, dcWk_b + w, dcbk + i*dm,
              dcWv_b + w, dcbv + i*dm, dcWo_b + w, dcbo + i*dm,
              dln2g + i*dm, dln2b + i*dm);
    ffn(dec_f, dec_b, B*512, dW1_b + wf1, db1 + i*dff, dW2_b + wf2, db2 + i*dm,
        dln3g + i*dm, dln3b + i*dm);
  }
  k_ln<<<B*512,128,0,stream>>>(dec_f, nullptr, decg, decb, nullptr, dm);
  { long long rows = (long long)B*L;
    k_proj<<<(unsigned)cdiv_ll(rows,256),256,0,stream>>>(dec_f, pW, pbv, (float*)d_out, dm, rows); }
}
